// MultiHeadAttention_12764642804020
// MI455X (gfx1250) — compile-verified, matrix operand order checked by reading, not run on hardware
//
#include <hip/hip_runtime.h>
#include <hip/hip_bf16.h>

// ---------------------------------------------------------------------------
// MI455X (gfx1250) multi-head causal attention, bf16 WMMA pipeline.
// wave32, v_wmma_f32_16x16x32_bf16, f32 accumulate.
// Unroll-by-2 ping-pong register pipelining (no rotation movs).
// ---------------------------------------------------------------------------

typedef __bf16 bf16_t;
typedef __attribute__((ext_vector_type(16))) __bf16 v16bf;
typedef __attribute__((ext_vector_type(8)))  __bf16 v8bf;
typedef __attribute__((ext_vector_type(8)))  float  v8f;

#define B_ 2
#define S_ 2048
#define E_ 768
#define H_ 12
#define D_ 64

union Frag16 { v16bf v; v8bf h[2]; };

__device__ __forceinline__ v8f wmma_bf16(v16bf a, v16bf b, v8f c) {
  // 8 args: (neg_a, A, neg_b, B, c_mod, C, reuse_a, reuse_b)
  return __builtin_amdgcn_wmma_f32_16x16x32_bf16(false, a, false, b, (short)0, c,
                                                 false, false);
}

// Load one 16x32 bf16 fragment in the CDNA5 16-bit A-matrix layout from a
// row-major tile (origin p, row stride ld elements, ld % 8 == 0 keeps the two
// 16-byte loads aligned).  Per ISA 7.12.2:
//   lanes 0-15  : row M=lane,    K = 0..7  (h[0]) and 16..23 (h[1])
//   lanes 16-31 : row M=lane-16, K = 8..15 (h[0]) and 24..31 (h[1])
// B-operands are fed as their transpose (N-major): same layout with M<->N.
__device__ __forceinline__ v16bf load_frag(const bf16_t* __restrict__ p, int ld) {
  const int lane = threadIdx.x & 31;
  const int r  = lane & 15;
  const int hh = lane >> 4;
  const bf16_t* base = p + (long)r * ld + hh * 8;
  Frag16 f;
  f.h[0] = *(const v8bf*)(base);
  f.h[1] = *(const v8bf*)(base + 16);
  return f.v;
}

// ---------------------------------------------------------------------------
// Precision conversion / layout kernels
// ---------------------------------------------------------------------------
__global__ void cast_bf16_kernel(const float* __restrict__ src,
                                 bf16_t* __restrict__ dst, int n) {
  for (int i = blockIdx.x * blockDim.x + threadIdx.x; i < n;
       i += gridDim.x * blockDim.x)
    dst[i] = (bf16_t)src[i];
}

// dst[mat][c][r] = (bf16)src[mat][r][c]
__global__ void transpose_cast_kernel(const float* __restrict__ src,
                                      bf16_t* __restrict__ dst,
                                      int rows, int cols, int nmat) {
  const long per = (long)rows * cols;
  const long total = per * nmat;
  for (long i = (long)blockIdx.x * blockDim.x + threadIdx.x; i < total;
       i += (long)gridDim.x * blockDim.x) {
    long mat = i / per;
    long rem = i - mat * per;
    int r = (int)(rem / cols);
    int c = (int)(rem - (long)r * cols);
    dst[mat * per + (long)c * rows + r] = (bf16_t)src[i];
  }
}

// ---------------------------------------------------------------------------
// QKV projection: per head  [S,E] @ [E,D] -> [S,D]  (bf16 in, bf16 out)
// Wt pre-transposed [H,D,E].  Q pre-scaled by 1/sqrt(D).  V written
// transposed [B,H,D,S] so it feeds the P*V WMMA B-operand directly.
// K loop unrolled x2 with ping-pong fragment registers.
// grid: (S/64, H, B*3)  block: 128 (4 waves, one 16x64 tile each)
// ---------------------------------------------------------------------------
__global__ void __launch_bounds__(128)
qkv_proj_kernel(const bf16_t* __restrict__ xb,
                const bf16_t* __restrict__ Wqt,
                const bf16_t* __restrict__ Wkt,
                const bf16_t* __restrict__ Wvt,
                bf16_t* __restrict__ Qb,
                bf16_t* __restrict__ Kb,
                bf16_t* __restrict__ Vt) {
  const int lane = threadIdx.x & 31;
  const int wave = threadIdx.x >> 5;
  const int which = blockIdx.z % 3;      // 0=Q 1=K 2=V
  const int b = blockIdx.z / 3;
  const int h = blockIdx.y;
  const int row0 = (blockIdx.x * 4 + wave) * 16;

  const bf16_t* Wt = (which == 0) ? Wqt : (which == 1) ? Wkt : Wvt;
  const bf16_t* Wh = Wt + (long)h * D_ * E_;
  const bf16_t* A0 = xb + ((long)b * S_ + row0) * E_;

  // ping set (ks = 0)
  v16bf a0 = load_frag(A0, E_);
  v16bf b0[4];
#pragma unroll
  for (int nd = 0; nd < 4; ++nd)
    b0[nd] = load_frag(Wh + (long)(nd * 16) * E_, E_);

  v8f acc[4] = {};
  for (int ks = 0; ks < E_; ks += 64) {
    // pong set (ks+32) loads issue before consuming the ping set
    v16bf a1 = load_frag(A0 + ks + 32, E_);
    v16bf b1[4];
#pragma unroll
    for (int nd = 0; nd < 4; ++nd)
      b1[nd] = load_frag(Wh + (long)(nd * 16) * E_ + ks + 32, E_);
#pragma unroll
    for (int nd = 0; nd < 4; ++nd)
      acc[nd] = wmma_bf16(a0, b0[nd], acc[nd]);
    // ping set for next iteration (ks+64); tail read overruns <2KB into the
    // next workspace array (layout-guarded).
    a0 = load_frag(A0 + ks + 64, E_);
#pragma unroll
    for (int nd = 0; nd < 4; ++nd)
      b0[nd] = load_frag(Wh + (long)(nd * 16) * E_ + ks + 64, E_);
#pragma unroll
    for (int nd = 0; nd < 4; ++nd)
      acc[nd] = wmma_bf16(a1, b1[nd], acc[nd]);
  }

  const int n_lo = lane & 15;
  const int hh = lane >> 4;
  const long bh = (long)b * H_ + h;
  if (which == 2) {
    // store V transposed: Vt[bh][d][s]
#pragma unroll
    for (int nd = 0; nd < 4; ++nd) {
      const int d = nd * 16 + n_lo;
      bf16_t* dst = Vt + (bh * D_ + d) * S_ + row0;
#pragma unroll
      for (int i = 0; i < 8; ++i)
        dst[hh * 8 + i] = (bf16_t)acc[nd][i];
    }
  } else {
    bf16_t* dst = (which == 0 ? Qb : Kb) + (bh * S_ + row0) * D_;
    const float scl = (which == 0) ? 0.125f : 1.0f;  // 1/sqrt(64) folded into Q
#pragma unroll
    for (int nd = 0; nd < 4; ++nd) {
#pragma unroll
      for (int i = 0; i < 8; ++i) {
        const int m = hh * 8 + i;
        dst[(long)m * D_ + nd * 16 + n_lo] = (bf16_t)(acc[nd][i] * scl);
      }
    }
  }
}

// ---------------------------------------------------------------------------
// Flash attention helper: one 32-key half-block of causal online softmax +
// P.V accumulation.  Scores arrive in WMMA C-layout (lane = col n_lo/+16,
// vgpr i = row hh*8+i); each score row lives across a 16-lane half-group.
// ---------------------------------------------------------------------------
__device__ __forceinline__ void softmax_pv(
    int kb, int row0, int n_lo, int hh, int wave,
    v8f s0, v8f s1, v8f* acc, float* rowM, float* rowL,
    const v16bf* vf, bf16_t* pbuf) {
#pragma unroll
  for (int i = 0; i < 8; ++i) {
    const int gq = row0 + hh * 8 + i;
    float v0 = s0[i], v1 = s1[i];
    if (kb + n_lo      > gq) v0 = -1e30f;
    if (kb + 16 + n_lo > gq) v1 = -1e30f;

    float mb = fmaxf(v0, v1);
    mb = fmaxf(mb, __shfl_xor(mb, 1, 32));
    mb = fmaxf(mb, __shfl_xor(mb, 2, 32));
    mb = fmaxf(mb, __shfl_xor(mb, 4, 32));
    mb = fmaxf(mb, __shfl_xor(mb, 8, 32));

    const float Mn = fmaxf(rowM[i], mb);
    const float alpha = __expf(rowM[i] - Mn);
    const float p0 = __expf(v0 - Mn);
    const float p1 = __expf(v1 - Mn);
    float ls = p0 + p1;
    ls += __shfl_xor(ls, 1, 32);
    ls += __shfl_xor(ls, 2, 32);
    ls += __shfl_xor(ls, 4, 32);
    ls += __shfl_xor(ls, 8, 32);
    rowL[i] = rowL[i] * alpha + ls;
    rowM[i] = Mn;
#pragma unroll
    for (int nd = 0; nd < 4; ++nd) acc[nd][i] *= alpha;

    // stage P tile to LDS in plain row-major [m][n]  (row stride 40 bf16)
    const int m = hh * 8 + i;
    pbuf[(wave * 16 + m) * 40 + n_lo]      = (bf16_t)p0;
    pbuf[(wave * 16 + m) * 40 + 16 + n_lo] = (bf16_t)p1;
  }
  // reload P in A-layout (per-wave region; in-wave DS ordering covers RAW)
  const v16bf ap = load_frag(pbuf + (long)wave * 16 * 40, 40);
#pragma unroll
  for (int nd = 0; nd < 4; ++nd)
    acc[nd] = wmma_bf16(ap, vf[nd], acc[nd]);
}

// ---------------------------------------------------------------------------
// Flash attention (causal, online softmax).  One wave per 16-row query block,
// two 32-key blocks per iteration with ping-pong K fragment sets.  K/V are
// L2-resident (12.6 MB vs 192 MB L2) so fragments come straight from global;
// each half issues the other half's K fragments and its own V fragments
// before its softmax so loads fly under the VALU work.  LDS is used only to
// move the probability tile from WMMA C-layout to A-layout.
// grid: (S/64, H, B)  block: 128
// ---------------------------------------------------------------------------
__global__ void __launch_bounds__(128)
flash_attn_kernel(const bf16_t* __restrict__ Qb,
                  const bf16_t* __restrict__ Kb,
                  const bf16_t* __restrict__ Vt,
                  bf16_t* __restrict__ concat) {
  __shared__ bf16_t pbuf[4 * 16 * 40];

  const int lane = threadIdx.x & 31;
  const int wave = threadIdx.x >> 5;
  const int b = blockIdx.z;
  const int h = blockIdx.y;
  const int row0 = (blockIdx.x * 4 + wave) * 16;
  const long bh = (long)b * H_ + h;
  const int n_lo = lane & 15;
  const int hh = lane >> 4;

  const bf16_t* Qt  = Qb + (bh * S_ + row0) * D_;
  const bf16_t* Kbh = Kb + bh * S_ * D_;
  const bf16_t* Vbh = Vt + bh * D_ * S_;
  const v16bf aq0 = load_frag(Qt + 0, D_);   // D = 0..31
  const v16bf aq1 = load_frag(Qt + 32, D_);  // D = 32..63

  v8f acc[4] = {};
  float rowM[8], rowL[8];
#pragma unroll
  for (int i = 0; i < 8; ++i) { rowM[i] = -1e30f; rowL[i] = 0.0f; }

  // ping set: K fragments for key block 0
  v16bf kfA0 = load_frag(Kbh, D_);
  v16bf kfA1 = load_frag(Kbh + 32, D_);
  v16bf kfA2 = load_frag(Kbh + 16 * D_, D_);
  v16bf kfA3 = load_frag(Kbh + 16 * D_ + 32, D_);

  const int nblk2 = (row0 + 16 + 63) / 64;    // 64-key double blocks (causal)
  for (int kb = 0; kb < nblk2 * 64; kb += 64) {
    // ---- half 0: keys [kb, kb+32) ----
    v8f s0 = {}, s1 = {};
    s0 = wmma_bf16(aq0, kfA0, s0);
    s0 = wmma_bf16(aq1, kfA1, s0);
    s1 = wmma_bf16(aq0, kfA2, s1);
    s1 = wmma_bf16(aq1, kfA3, s1);

    const bf16_t* Kh1 = Kbh + (long)(kb + 32) * D_;   // pong set loads
    v16bf kfB0 = load_frag(Kh1, D_);
    v16bf kfB1 = load_frag(Kh1 + 32, D_);
    v16bf kfB2 = load_frag(Kh1 + 16 * D_, D_);
    v16bf kfB3 = load_frag(Kh1 + 16 * D_ + 32, D_);
    v16bf vf[4];
#pragma unroll
    for (int nd = 0; nd < 4; ++nd)
      vf[nd] = load_frag(Vbh + (long)(nd * 16) * S_ + kb, S_);

    // deep prefetch, two double-blocks ahead (global_prefetch_b8; speculative
    // prefetches past the buffer are silently dropped).
    {
      const char* pk = (const char*)(Kbh + (long)(kb + 128) * D_);
      __builtin_prefetch(pk + lane * 128, 0, 3);             // 4KB K tile
      const char* pk2 = (const char*)(Kbh + (long)(kb + 160) * D_);
      __builtin_prefetch(pk2 + lane * 128, 0, 3);            // next 4KB K tile
      const bf16_t* pv = Vbh + (kb + 128);
      __builtin_prefetch(pv + (long)lane * S_, 0, 3);        // V rows 0..31
      __builtin_prefetch(pv + (long)(lane + 32) * S_, 0, 3); // V rows 32..63
    }

    softmax_pv(kb, row0, n_lo, hh, wave, s0, s1, acc, rowM, rowL, vf, pbuf);

    // ---- half 1: keys [kb+32, kb+64) ----
    v8f t0 = {}, t1 = {};
    t0 = wmma_bf16(aq0, kfB0, t0);
    t0 = wmma_bf16(aq1, kfB1, t0);
    t1 = wmma_bf16(aq0, kfB2, t1);
    t1 = wmma_bf16(aq1, kfB3, t1);

    const bf16_t* Kh2 = Kbh + (long)(kb + 64) * D_;   // ping set for next iter
    kfA0 = load_frag(Kh2, D_);
    kfA1 = load_frag(Kh2 + 32, D_);
    kfA2 = load_frag(Kh2 + 16 * D_, D_);
    kfA3 = load_frag(Kh2 + 16 * D_ + 32, D_);
#pragma unroll
    for (int nd = 0; nd < 4; ++nd)
      vf[nd] = load_frag(Vbh + (long)(nd * 16) * S_ + kb + 32, S_);

    softmax_pv(kb + 32, row0, n_lo, hh, wave, t0, t1, acc, rowM, rowL, vf, pbuf);
  }

  // normalize and write concat [B, S, H*D] (bf16 for the output projection)
  bf16_t* dst = concat + ((long)b * S_ + row0) * (H_ * D_) + h * D_;
#pragma unroll
  for (int i = 0; i < 8; ++i) {
    const float inv = 1.0f / rowL[i];
    const int m = hh * 8 + i;
#pragma unroll
    for (int nd = 0; nd < 4; ++nd)
      dst[(long)m * (H_ * D_) + nd * 16 + n_lo] = (bf16_t)(acc[nd][i] * inv);
  }
}

// ---------------------------------------------------------------------------
// Output projection: [B*S, 768] @ Wo[768,768] + bo -> f32 out
// Wot is Wo^T (row-major [n][k]).  K loop unrolled x2, ping-pong registers.
// grid: (B*S/64, E/64)  block: 128
// ---------------------------------------------------------------------------
__global__ void __launch_bounds__(128)
out_proj_kernel(const bf16_t* __restrict__ concat,
                const bf16_t* __restrict__ Wot,
                const float* __restrict__ bo,
                float* __restrict__ out) {
  const int lane = threadIdx.x & 31;
  const int wave = threadIdx.x >> 5;
  const int row0 = (blockIdx.x * 4 + wave) * 16;  // over B*S rows
  const int ncol0 = blockIdx.y * 64;

  const bf16_t* A0 = concat + (long)row0 * E_;
  const bf16_t* W0 = Wot + (long)ncol0 * E_;

  v16bf a0 = load_frag(A0, E_);
  v16bf b0[4];
#pragma unroll
  for (int nd = 0; nd < 4; ++nd)
    b0[nd] = load_frag(W0 + (long)(nd * 16) * E_, E_);

  v8f acc[4] = {};
  for (int ks = 0; ks < E_; ks += 64) {
    v16bf a1 = load_frag(A0 + ks + 32, E_);
    v16bf b1[4];
#pragma unroll
    for (int nd = 0; nd < 4; ++nd)
      b1[nd] = load_frag(W0 + (long)(nd * 16) * E_ + ks + 32, E_);
#pragma unroll
    for (int nd = 0; nd < 4; ++nd)
      acc[nd] = wmma_bf16(a0, b0[nd], acc[nd]);
    a0 = load_frag(A0 + ks + 64, E_);
#pragma unroll
    for (int nd = 0; nd < 4; ++nd)
      b0[nd] = load_frag(W0 + (long)(nd * 16) * E_ + ks + 64, E_);
#pragma unroll
    for (int nd = 0; nd < 4; ++nd)
      acc[nd] = wmma_bf16(a1, b1[nd], acc[nd]);
  }

  const int n_lo = lane & 15;
  const int hh = lane >> 4;
#pragma unroll
  for (int nd = 0; nd < 4; ++nd) {
    const int n = ncol0 + nd * 16 + n_lo;
    const float bias = bo[n];
#pragma unroll
    for (int i = 0; i < 8; ++i) {
      const int m = hh * 8 + i;
      out[(long)(row0 + m) * E_ + n] = acc[nd][i] + bias;
    }
  }
}

// ---------------------------------------------------------------------------
// Host launcher
// ---------------------------------------------------------------------------
extern "C" void kernel_launch(void* const* d_in, const int* in_sizes, int n_in,
                              void* d_out, int out_size, void* d_ws, size_t ws_size,
                              hipStream_t stream) {
  (void)in_sizes; (void)n_in; (void)out_size; (void)ws_size;
  const float* x  = (const float*)d_in[0];   // [B,S,E]
  const float* Wq = (const float*)d_in[1];   // [H,E,D]
  const float* Wk = (const float*)d_in[2];
  const float* Wv = (const float*)d_in[3];
  const float* Wo = (const float*)d_in[4];   // [H*D,E]
  const float* bo = (const float*)d_in[5];   // [E]
  float* out = (float*)d_out;                // [B,S,E] f32

  // Layout ordered so every pipelined tail-read overrun (<16KB) lands inside
  // the NEXT workspace array, never past the end of d_ws:
  //   xb -> Wqt -> Wkt -> Wvt -> Wot -> concat -> Qb -> Kb -> Vt
  char* ws = (char*)d_ws;
  size_t off = 0;
  auto alloc = [&](size_t bytes) -> void* {
    void* p = ws + off;
    off = (off + bytes + 255) & ~(size_t)255;
    return p;
  };
  bf16_t* xb     = (bf16_t*)alloc((size_t)B_ * S_ * E_ * 2);      // x in bf16
  bf16_t* Wqt    = (bf16_t*)alloc((size_t)H_ * D_ * E_ * 2);      // [H,D,E]
  bf16_t* Wkt    = (bf16_t*)alloc((size_t)H_ * D_ * E_ * 2);
  bf16_t* Wvt    = (bf16_t*)alloc((size_t)H_ * D_ * E_ * 2);
  bf16_t* Wot    = (bf16_t*)alloc((size_t)E_ * E_ * 2);           // Wo^T
  bf16_t* concat = (bf16_t*)alloc((size_t)B_ * S_ * H_ * D_ * 2); // [B,S,H*D]
  bf16_t* Qb     = (bf16_t*)alloc((size_t)B_ * H_ * S_ * D_ * 2); // [B,H,S,D]
  bf16_t* Kb     = (bf16_t*)alloc((size_t)B_ * H_ * S_ * D_ * 2);
  bf16_t* Vt     = (bf16_t*)alloc((size_t)B_ * H_ * D_ * S_ * 2); // [B,H,D,S]

  cast_bf16_kernel<<<1024, 256, 0, stream>>>(x, xb, B_ * S_ * E_);
  transpose_cast_kernel<<<512, 256, 0, stream>>>(Wq, Wqt, E_, D_, H_);
  transpose_cast_kernel<<<512, 256, 0, stream>>>(Wk, Wkt, E_, D_, H_);
  transpose_cast_kernel<<<512, 256, 0, stream>>>(Wv, Wvt, E_, D_, H_);
  transpose_cast_kernel<<<512, 256, 0, stream>>>(Wo, Wot, H_ * D_, E_, 1);

  qkv_proj_kernel<<<dim3(S_ / 64, H_, B_ * 3), 128, 0, stream>>>(
      xb, Wqt, Wkt, Wvt, Qb, Kb, Vt);

  flash_attn_kernel<<<dim3(S_ / 64, H_, B_), 128, 0, stream>>>(
      Qb, Kb, Vt, concat);

  out_proj_kernel<<<dim3((B_ * S_) / 64, E_ / 64), 128, 0, stream>>>(
      concat, Wot, bo, out);
}
